// MultiheadAttention_test_35725537968476
// MI455X (gfx1250) — compile-verified, matrix operand order checked by reading, not run on hardware
//
#include <hip/hip_runtime.h>
#include <hip/hip_bf16.h>

// ---------------------------------------------------------------------------
// Problem constants (match reference: B=2, T=2048, D=1024, HEADS=16)
// ---------------------------------------------------------------------------
constexpr int BB   = 2;
constexpr int TT   = 2048;
constexpr int DD   = 1024;
constexpr int HH   = 16;
constexpr int HD   = DD / HH;        // 64
constexpr int MM   = BB * TT;        // 4096 rows of X
constexpr int N3   = 3 * DD;         // 3072

// ---------------------------------------------------------------------------
// WMMA types / helpers (CDNA5 wave32, V_WMMA_F32_16X16X32_BF16)
// ---------------------------------------------------------------------------
typedef __attribute__((ext_vector_type(16))) __bf16 bf16x16;
typedef __attribute__((ext_vector_type(8)))  float  floatx8;

__device__ __forceinline__ floatx8 wmma_bf16(bf16x16 a, bf16x16 b, floatx8 c) {
  // (neg_a, A, neg_b, B, c_mod, C, reuse_a, reuse_b)
  return __builtin_amdgcn_wmma_f32_16x16x32_bf16(false, a, false, b,
                                                 (short)0, c, false, false);
}

__device__ __forceinline__ unsigned short f32_to_bf16_rne(float f) {
  unsigned int u = __float_as_uint(f);
  u += 0x7fffu + ((u >> 16) & 1u);   // round to nearest even
  return (unsigned short)(u >> 16);
}

__device__ __forceinline__ __bf16 us_as_bf16(unsigned short u) {
  return __builtin_bit_cast(__bf16, u);
}

// ISA 05_wmma §7.12.2: 16-bit A 16x32 — lane holds row M=lane&15,
// half = lane>>4; VGPR v holds K = kbase(v,half), kbase(v,half)+1.
__device__ __forceinline__ int frag_kbase(int v, int half) {
  int base = (v < 4) ? (v * 2) : (16 + (v - 4) * 2);
  return base + half * 8;
}

// Fragment from a row-major bf16 matrix where the WMMA K-dim is the
// contiguous dim: elem(r, k) = base[r*stride + k].  Serves as:
//   * A operand (r = m)
//   * B operand when memory holds B^T row-major (r = n, k contiguous)
// Lowered to two global_load_b128 per lane (16B chunks at k=8h.. and 16+8h..).
__device__ __forceinline__ bf16x16
load_frag_pairs(const unsigned short* base, long long stride, int lane) {
  int r = lane & 15, half = lane >> 4;
  const unsigned short* row = base + (long long)r * stride;
  bf16x16 f;
#pragma unroll
  for (int v = 0; v < 8; ++v) {
    int k = frag_kbase(v, half);                       // always even
    unsigned int pair = *(const unsigned int*)(row + k);
    f[2 * v]     = us_as_bf16((unsigned short)(pair & 0xffffu));
    f[2 * v + 1] = us_as_bf16((unsigned short)(pair >> 16));
  }
  return f;
}

// Same fragment but source is f32 (attention probabilities), converted
// to bf16 in-register.
__device__ __forceinline__ bf16x16
load_frag_pairs_f32(const float* base, long long stride, int lane) {
  int r = lane & 15, half = lane >> 4;
  const float* row = base + (long long)r * stride;
  bf16x16 f;
#pragma unroll
  for (int v = 0; v < 8; ++v) {
    int k = frag_kbase(v, half);
    f[2 * v]     = us_as_bf16(f32_to_bf16_rne(row[k]));
    f[2 * v + 1] = us_as_bf16(f32_to_bf16_rne(row[k + 1]));
  }
  return f;
}

// B fragment from an LDS-staged B^T tile (row stride LDS_STRIDE elems,
// padded so the 16 lanes' ds_load_b128 reads land on distinct bank sets).
constexpr int LDS_STRIDE = 40;   // 80 bytes; 20*n mod 64 distinct for n=0..15
__device__ __forceinline__ bf16x16
load_frag_lds(const unsigned short* tile, int j, int lane) {
  int n = lane & 15, half = lane >> 4;
  const unsigned short* row = tile + (j * 16 + n) * LDS_STRIDE + half * 8;
  union { uint4 q[2]; bf16x16 f; } u;
  u.q[0] = *(const uint4*)(row);        // k =      8h .. 8h+7   (v = 0..3)
  u.q[1] = *(const uint4*)(row + 16);   // k = 16 + 8h .. +7     (v = 4..7)
  return u.f;
}

// ---------------------------------------------------------------------------
// Kernel: f32 -> bf16 (stored as u16) elementwise convert
// ---------------------------------------------------------------------------
__global__ __launch_bounds__(256) void convert_f32_bf16(const float* __restrict__ in,
                                                        unsigned short* __restrict__ out,
                                                        int n) {
  int i = blockIdx.x * 256 + threadIdx.x;
  if (i < n) out[i] = f32_to_bf16_rne(in[i]);
}

// Kernel: f32 [K x N] -> bf16 transposed [N x K] (weights become N-major so
// WMMA B fragments are contiguous-pair loads).  i is linear over the output.
__global__ __launch_bounds__(256) void convert_transpose_f32_bf16(
    const float* __restrict__ in, unsigned short* __restrict__ out,
    int Kdim, int Ncols) {
  int i = blockIdx.x * 256 + threadIdx.x;
  if (i >= Kdim * Ncols) return;
  int k = i & (Kdim - 1);              // Kdim is a power of two (1024)
  int n = i >> 10;
  out[(long long)n * Kdim + k] = f32_to_bf16_rne(in[(long long)k * Ncols + n]);
}

// Kernel: build Vt[bh][d][t] (bf16) from KQV rows: V block cols 2D..3D-1.
__global__ __launch_bounds__(256) void transpose_v(
    const unsigned short* __restrict__ KQV, unsigned short* __restrict__ Vt) {
  long long i = (long long)blockIdx.x * 256 + threadIdx.x;   // over BH*HD*TT
  int t  = (int)(i & (TT - 1));
  int d  = (int)((i >> 11) & (HD - 1));
  int bh = (int)(i >> 17);
  int b = bh >> 4, h = bh & 15;
  Vt[i] = KQV[((long long)b * TT + t) * N3 + 2 * DD + h * HD + d];
}

// ---------------------------------------------------------------------------
// Kernel: C[M,N] = A[M,K] @ B[K,N] with B supplied transposed (Bt = [N,K]).
// Block = 256 threads = 8 waves; wave w owns rows [by*128 + w*16, +16),
// cols [bx*64, +64).  The 64x32 Bt tile for the current K-step is shared by
// all 8 waves, so it is staged into LDS once per step with
// global_load_async_to_lds_b128 (ASYNCcnt path), then read back as
// ds_load_b128 fragments.
// ---------------------------------------------------------------------------
template <bool BF16_OUT>
__global__ __launch_bounds__(256) void gemm_wmma(const unsigned short* __restrict__ A,
                                                 const unsigned short* __restrict__ Bt,
                                                 void* __restrict__ Cout,
                                                 int Kdim, int Ncols) {
  __shared__ unsigned short btile[64 * LDS_STRIDE];

  const int lane = threadIdx.x & 31;
  const int wave = threadIdx.x >> 5;
  const long long m0 = (long long)blockIdx.y * 128 + wave * 16;
  const long long n0 = (long long)blockIdx.x * 64;

  // Staging map: 256 threads x 16B = 64 rows x 64B (32 bf16 of K).
  const int srow = threadIdx.x >> 2;                 // 0..63
  const int sseg = threadIdx.x & 3;                  // 0..3 (8 elems each)
  const unsigned lds_dst =
      (unsigned)(uintptr_t)&btile[srow * LDS_STRIDE + sseg * 8];
  const unsigned short* gsrc =
      Bt + (n0 + srow) * (long long)Kdim + sseg * 8;

  floatx8 acc0 = {}, acc1 = {}, acc2 = {}, acc3 = {};
  const unsigned short* Arow = A + m0 * Kdim;

  for (int k0 = 0; k0 < Kdim; k0 += 32) {
    // Async-stage Bt rows [n0, n0+64) x cols [k0, k0+32) into LDS.
    {
      unsigned long long ga = (unsigned long long)(uintptr_t)(gsrc + k0);
      asm volatile("global_load_async_to_lds_b128 %0, %1, off"
                   :: "v"(lds_dst), "v"(ga) : "memory");
      asm volatile("s_wait_asynccnt 0x0" ::: "memory");
    }
    __syncthreads();

    __builtin_prefetch(Arow + k0 + 256, 0, 0);       // global_prefetch_b8
    bf16x16 a  = load_frag_pairs(Arow + k0, Kdim, lane);
    bf16x16 b0 = load_frag_lds(btile, 0, lane);
    bf16x16 b1 = load_frag_lds(btile, 1, lane);
    bf16x16 b2 = load_frag_lds(btile, 2, lane);
    bf16x16 b3 = load_frag_lds(btile, 3, lane);
    acc0 = wmma_bf16(a, b0, acc0);
    acc1 = wmma_bf16(a, b1, acc1);
    acc2 = wmma_bf16(a, b2, acc2);
    acc3 = wmma_bf16(a, b3, acc3);
    __syncthreads();
  }

  const int n = lane & 15, half = lane >> 4;
#pragma unroll
  for (int v = 0; v < 8; ++v) {
    long long row = m0 + v + 8 * half;
    if (BF16_OUT) {
      unsigned short* C = (unsigned short*)Cout + row * Ncols + n0 + n;
      C[0]  = f32_to_bf16_rne(acc0[v]);
      C[16] = f32_to_bf16_rne(acc1[v]);
      C[32] = f32_to_bf16_rne(acc2[v]);
      C[48] = f32_to_bf16_rne(acc3[v]);
    } else {
      float* C = (float*)Cout + row * Ncols + n0 + n;
      C[0] = acc0[v]; C[16] = acc1[v]; C[32] = acc2[v]; C[48] = acc3[v];
    }
  }
}

// ---------------------------------------------------------------------------
// Kernel: scores[b,h,k,q] = (K_h[k,:] . Q_h[q,:]) / 8 + mask[k,q]
// (reference computes K @ Q^T — reproduced faithfully).  Wave = 16 k-rows x
// 64 q-cols: one K fragment feeds 4 WMMAs.  B = Q^T with Q row-major, so B
// fragments are contiguous-pair loads.
// ---------------------------------------------------------------------------
__global__ __launch_bounds__(128) void attn_scores(const unsigned short* __restrict__ KQV,
                                                   const float* __restrict__ mask,
                                                   float* __restrict__ attn) {
  const int lane = threadIdx.x & 31;
  const int wave = threadIdx.x >> 5;                 // 0..3
  const int bh = blockIdx.z, b = bh >> 4, h = bh & 15;
  const long long k0 = (long long)blockIdx.x * 64 + wave * 16;
  const long long q0 = (long long)blockIdx.y * 64;

  const unsigned short* Kb = KQV + ((long long)b * TT) * N3 + h * HD;
  const unsigned short* Qb = Kb + DD;                // Q block offset

  floatx8 acc0 = {}, acc1 = {}, acc2 = {}, acc3 = {};
#pragma unroll
  for (int d0 = 0; d0 < HD; d0 += 32) {
    bf16x16 a  = load_frag_pairs(Kb + k0 * N3 + d0, N3, lane);
    bf16x16 q1 = load_frag_pairs(Qb + (q0 +  0) * N3 + d0, N3, lane);
    bf16x16 q2 = load_frag_pairs(Qb + (q0 + 16) * N3 + d0, N3, lane);
    bf16x16 q3 = load_frag_pairs(Qb + (q0 + 32) * N3 + d0, N3, lane);
    bf16x16 q4 = load_frag_pairs(Qb + (q0 + 48) * N3 + d0, N3, lane);
    acc0 = wmma_bf16(a, q1, acc0);
    acc1 = wmma_bf16(a, q2, acc1);
    acc2 = wmma_bf16(a, q3, acc2);
    acc3 = wmma_bf16(a, q4, acc3);
  }

  float* out = attn + ((long long)bh * TT + k0) * TT + q0;
  const float* mrow = mask + k0 * TT + q0;
  const int n = lane & 15, half = lane >> 4;
#pragma unroll
  for (int v = 0; v < 8; ++v) {
    long long r = v + 8 * half;
    out[r * TT + n +  0] = acc0[v] * 0.125f + mrow[r * TT + n +  0];
    out[r * TT + n + 16] = acc1[v] * 0.125f + mrow[r * TT + n + 16];
    out[r * TT + n + 32] = acc2[v] * 0.125f + mrow[r * TT + n + 32];
    out[r * TT + n + 48] = acc3[v] * 0.125f + mrow[r * TT + n + 48];
  }
}

// ---------------------------------------------------------------------------
// Kernel: in-place row softmax over the q axis (T=2048 cols per row).
// ---------------------------------------------------------------------------
__global__ __launch_bounds__(256) void softmax_rows(float* __restrict__ attn) {
  __shared__ float red[256];
  float* p = attn + (long long)blockIdx.x * TT;
  const int tid = threadIdx.x;

  float m = -3.0e38f;
  for (int i = tid; i < TT; i += 256) m = fmaxf(m, p[i]);
  red[tid] = m; __syncthreads();
  for (int s = 128; s > 0; s >>= 1) {
    if (tid < s) red[tid] = fmaxf(red[tid], red[tid + s]);
    __syncthreads();
  }
  m = red[0]; __syncthreads();

  float sum = 0.f;
  for (int i = tid; i < TT; i += 256) {
    float e = __expf(p[i] - m);
    p[i] = e;
    sum += e;
  }
  red[tid] = sum; __syncthreads();
  for (int s = 128; s > 0; s >>= 1) {
    if (tid < s) red[tid] += red[tid + s];
    __syncthreads();
  }
  const float inv = 1.0f / red[0];
  for (int i = tid; i < TT; i += 256) p[i] *= inv;
}

// ---------------------------------------------------------------------------
// Kernel: ctx[b,t, h*64+d] = sum_q attn[b,h,t,q] * V[b,h,q,d]
// Wave = 16 t-rows x full head dim (64): one attn fragment feeds 4 WMMAs, so
// the 537 MB attn tensor is read exactly once.  B comes from Vt[bh][d][t]
// (t contiguous) -> contiguous-pair loads.
// ---------------------------------------------------------------------------
__global__ __launch_bounds__(256) void attn_ctx(const float* __restrict__ attn,
                                                const unsigned short* __restrict__ Vt,
                                                unsigned short* __restrict__ ctx) {
  const int lane = threadIdx.x & 31;
  const int wave = threadIdx.x >> 5;
  const int bh = blockIdx.y, b = bh >> 4, h = bh & 15;
  const long long t0 = (long long)blockIdx.x * 128 + wave * 16;

  const float* Ab = attn + ((long long)bh * TT + t0) * TT;
  const unsigned short* Vb = Vt + (long long)bh * HD * TT;   // rows d, stride T

  floatx8 acc0 = {}, acc1 = {}, acc2 = {}, acc3 = {};
  for (int q0 = 0; q0 < TT; q0 += 32) {
    bf16x16 a  = load_frag_pairs_f32(Ab + q0, TT, lane);
    bf16x16 v0 = load_frag_pairs(Vb + (long long) 0 * TT + q0, TT, lane);
    bf16x16 v1 = load_frag_pairs(Vb + (long long)16 * TT + q0, TT, lane);
    bf16x16 v2 = load_frag_pairs(Vb + (long long)32 * TT + q0, TT, lane);
    bf16x16 v3 = load_frag_pairs(Vb + (long long)48 * TT + q0, TT, lane);
    acc0 = wmma_bf16(a, v0, acc0);
    acc1 = wmma_bf16(a, v1, acc1);
    acc2 = wmma_bf16(a, v2, acc2);
    acc3 = wmma_bf16(a, v3, acc3);
  }

  unsigned short* out = ctx + ((long long)b * TT + t0) * DD + h * HD;
  const int n = lane & 15, half = lane >> 4;
#pragma unroll
  for (int v = 0; v < 8; ++v) {
    long long r = v + 8 * half;
    out[r * DD + n +  0] = f32_to_bf16_rne(acc0[v]);
    out[r * DD + n + 16] = f32_to_bf16_rne(acc1[v]);
    out[r * DD + n + 32] = f32_to_bf16_rne(acc2[v]);
    out[r * DD + n + 48] = f32_to_bf16_rne(acc3[v]);
  }
}

// ---------------------------------------------------------------------------
// Host launcher
// ---------------------------------------------------------------------------
extern "C" void kernel_launch(void* const* d_in, const int* in_sizes, int n_in,
                              void* d_out, int out_size, void* d_ws, size_t ws_size,
                              hipStream_t stream) {
  (void)in_sizes; (void)n_in; (void)out_size; (void)ws_size;

  const float* X     = (const float*)d_in[0];   // [B,T,D]
  const float* W_KQV = (const float*)d_in[1];   // [D,3D]
  const float* W_out = (const float*)d_in[2];   // [D,D]
  const float* mask  = (const float*)d_in[3];   // [T,T]
  // d_in[4] = heads (scalar); constants match the reference.

  // d_out layout: out f32 [B*T*D] then attn f32 [B*H*T*T]
  float* out_f32  = (float*)d_out;
  float* attn_f32 = out_f32 + (long long)MM * DD;

  // Workspace layout (56 MB total)
  char* ws = (char*)d_ws;
  unsigned short* X_bf     = (unsigned short*)(ws);                  //  8 MB
  unsigned short* WkqvT_bf = (unsigned short*)(ws + (8ll  << 20));   //  6 MB [3072x1024]
  unsigned short* WoutT_bf = (unsigned short*)(ws + (14ll << 20));   //  2 MB [1024x1024]
  unsigned short* KQV_bf   = (unsigned short*)(ws + (16ll << 20));   // 24 MB [4096x3072]
  unsigned short* ctx_bf   = (unsigned short*)(ws + (40ll << 20));   //  8 MB [4096x1024]
  unsigned short* Vt_bf    = (unsigned short*)(ws + (48ll << 20));   //  8 MB [32x64x2048]

  // 1) conversions: X plain; weights converted AND transposed to [N,K]
  {
    int n = MM * DD;
    convert_f32_bf16<<<(n + 255) / 256, 256, 0, stream>>>(X, X_bf, n);
    n = DD * N3;
    convert_transpose_f32_bf16<<<(n + 255) / 256, 256, 0, stream>>>(
        W_KQV, WkqvT_bf, DD, N3);
    n = DD * DD;
    convert_transpose_f32_bf16<<<(n + 255) / 256, 256, 0, stream>>>(
        W_out, WoutT_bf, DD, DD);
  }

  // 2) KQV = X @ W_KQV   [4096x1024] x [1024x3072] -> bf16
  gemm_wmma<true><<<dim3(N3 / 64, MM / 128), 256, 0, stream>>>(
      X_bf, WkqvT_bf, (void*)KQV_bf, DD, N3);

  // 3) Vt[bh][d][t] transpose of the V block
  {
    long long n = (long long)BB * HH * HD * TT;   // 4M
    transpose_v<<<(unsigned)((n + 255) / 256), 256, 0, stream>>>(KQV_bf, Vt_bf);
  }

  // 4) scores = K @ Q^T / 8 + mask  -> attn region of d_out (f32)
  attn_scores<<<dim3(TT / 64, TT / 64, BB * HH), 128, 0, stream>>>(
      KQV_bf, mask, attn_f32);

  // 5) softmax over q, in place (B*H*T rows)
  softmax_rows<<<BB * HH * TT, 256, 0, stream>>>(attn_f32);

  // 6) ctx = attn @ V  -> bf16 [4096x1024]
  attn_ctx<<<dim3(TT / 128, BB * HH), 256, 0, stream>>>(attn_f32, Vt_bf, ctx_bf);

  // 7) out = ctx @ W_out  [4096x1024] x [1024x1024] -> f32 d_out
  gemm_wmma<false><<<dim3(DD / 64, MM / 128), 256, 0, stream>>>(
      ctx_bf, WoutT_bf, (void*)out_f32, DD, DD);
}
